// GlobalBlock_20710332301890
// MI455X (gfx1250) — matrix operand order repaired and checked by reading, not compiled, not run on hardware
//
#include <hip/hip_runtime.h>
#include <hip/hip_bf16.h>

// B=8, C=128, L=2048 attention block. ~19.3 GFLOP vs ~16MB HBM traffic ->
// matrix-core bound. Strategy: f16 WMMA (16x16x32, f32 accum) everywhere.
// Weights are pre-converted to f16 once so every B-fragment is one 32-byte load.

typedef _Float16 h4   __attribute__((ext_vector_type(4)));
typedef _Float16 h8   __attribute__((ext_vector_type(8)));
typedef _Float16 v16h __attribute__((ext_vector_type(16)));
typedef float    v8f  __attribute__((ext_vector_type(8)));

constexpr int B_ = 8;
constexpr int C_ = 128;
constexpr int L_ = 2048;
constexpr float SM_SCALE = 0.08838834764831845f;   // 1/sqrt(128)
constexpr float LOG2E    = 1.4426950408889634f;
constexpr float NEG_INF  = -1.0e30f;

static __device__ __forceinline__ v8f wmma_f16(v16h a, v16h b, v8f c) {
  // D = A(16x32 f16) x B(32x16 f16) + C(16x16 f32)
  return __builtin_amdgcn_wmma_f32_16x16x32_f16(
      /*neg_a=*/false, a, /*neg_b=*/false, b,
      /*c_mod=*/(short)0, c, /*reuse_a=*/false, /*reuse_b=*/false);
}

// B-fragment from row-major f16 weight W[o][c]: B[k=c][n=o] = W[o][c].
// Per lane: n = lane&15 (+tile), k(j) = 16*half + j -> 16 CONTIGUOUS halves:
// a single 32-byte load, no conversion.
static __device__ __forceinline__ v16h load_wfrag_h(
    const _Float16* __restrict__ W, int o, int c0) {
  return *reinterpret_cast<const v16h*>(W + (size_t)o * C_ + c0);
}

// Build an A-fragment from two contiguous 8-halfword runs (k order here is
// k(j) = 16*half + j, as in the B-frags -> runs at +16*half and +16*half+8).
static __device__ __forceinline__ v16h combine_a(h8 lo, h8 hi) {
  v16h r;
#pragma unroll
  for (int j = 0; j < 8; ++j) { r[j] = lo[j]; r[j + 8] = hi[j]; }
  return r;
}

// ---------------------------------------------------------------------------
// Kernel 0: one-shot f32 -> f16 conversion of the four 128x128 weights.
// ---------------------------------------------------------------------------
__global__ __launch_bounds__(256) void wcvt_kernel(
    const float* __restrict__ W0, const float* __restrict__ W1,
    const float* __restrict__ W2, const float* __restrict__ W3,
    _Float16* __restrict__ dst) {
  const float* src[4] = {W0, W1, W2, W3};
  const float* W = src[blockIdx.y];
  _Float16* d = dst + (size_t)blockIdx.y * C_ * C_;
  const int idx = (blockIdx.x * 256 + threadIdx.x) * 4;  // 16 blocks x 256 x 4
  const float4 f = *reinterpret_cast<const float4*>(W + idx);
  h4 o;
  o[0] = (_Float16)f.x; o[1] = (_Float16)f.y;
  o[2] = (_Float16)f.z; o[3] = (_Float16)f.w;
  *reinterpret_cast<h4*>(d + idx) = o;
}

// ---------------------------------------------------------------------------
// Kernel 1: h = relu(x); Q/K = [B,L,C] f16 ; V stored transposed [B,C,L] f16
// ---------------------------------------------------------------------------
__global__ __launch_bounds__(128) void qkv_proj_kernel(
    const float* __restrict__ x,
    const _Float16* __restrict__ Wqh, const float* __restrict__ bq,
    const _Float16* __restrict__ Wkh, const float* __restrict__ bk,
    const _Float16* __restrict__ Wvh, const float* __restrict__ bv,
    _Float16* __restrict__ Qw, _Float16* __restrict__ Kw,
    _Float16* __restrict__ Vw) {
  const int tid  = threadIdx.x;
  const int wave = tid >> 5, lane = tid & 31;
  const int half = lane >> 4, lr = lane & 15;
  const int blk  = blockIdx.x;
  const int b    = blk >> 5;            // 32 l-tiles per batch
  const int lbase = (blk & 31) * 64 + wave * 16;

  // A fragments: A[m=l][k=c] = relu(x[b][c][l]) (gather along c, stride L)
  v16h afrag[4];
  const int l = lbase + lr;
#pragma unroll
  for (int kc = 0; kc < 4; ++kc) {
#pragma unroll
    for (int j = 0; j < 16; ++j) {
      const int c = kc * 32 + 16 * half + j;
      const float xv = x[((size_t)b * C_ + c) * L_ + l];
      afrag[kc][j] = (_Float16)fmaxf(xv, 0.0f);
    }
  }

#pragma unroll 1
  for (int nt = 0; nt < 8; ++nt) {
    const int o = nt * 16 + lr;
    v8f accq = {}, acck = {}, accv = {};
#pragma unroll
    for (int kc = 0; kc < 4; ++kc) {
      const int c0 = kc * 32 + 16 * half;
      const v16h bqf = load_wfrag_h(Wqh, o, c0);
      const v16h bkf = load_wfrag_h(Wkh, o, c0);
      const v16h bvf = load_wfrag_h(Wvh, o, c0);
      accq = wmma_f16(afrag[kc], bqf, accq);
      acck = wmma_f16(afrag[kc], bkf, acck);
      accv = wmma_f16(afrag[kc], bvf, accv);
    }
    const float biq = bq[o], bik = bk[o], biv = bv[o];
    // Q,K -> [B,L,C]: D VGPR i holds row l=lbase+i+8*half, col o (per-lane)
#pragma unroll
    for (int i = 0; i < 8; ++i) {
      const int row = lbase + i + 8 * half;
      const size_t off = ((size_t)b * L_ + row) * C_ + o;
      Qw[off] = (_Float16)(accq[i] + biq);
      Kw[off] = (_Float16)(acck[i] + bik);
    }
    // V -> [B,C,L]: per lane the 8 rows are CONSECUTIVE l -> one b128 store
    h8 vpack;
#pragma unroll
    for (int i = 0; i < 8; ++i) vpack[i] = (_Float16)(accv[i] + biv);
    *reinterpret_cast<h8*>(Vw + ((size_t)b * C_ + o) * L_ + lbase + 8 * half) =
        vpack;
  }
}

// ---------------------------------------------------------------------------
// Kernel 2: flash attention. Per wave: 16 query rows, kv tiles of 32.
// R = relu(softmax(QK^T/sqrt(C)) V) stored f16 [B,L,C].
// ---------------------------------------------------------------------------
__global__ __launch_bounds__(128) void attn_kernel(
    const _Float16* __restrict__ Qw, const _Float16* __restrict__ Kw,
    const _Float16* __restrict__ Vw, _Float16* __restrict__ Rw) {
  __shared__ __align__(16) _Float16 lds_p[4][16][40];  // per-wave 16x32 P tile

  const int tid  = threadIdx.x;
  const int wave = tid >> 5, lane = tid & 31;
  const int half = lane >> 4, lr = lane & 15;
  const int blk  = blockIdx.x;
  const int b    = blk >> 5;
  const int lbase = (blk & 31) * 64 + wave * 16;

  // Resident Q fragments (A layout): row = lr, contiguous 16B runs in c.
  v16h qfrag[4];
  {
    const _Float16* qrow = Qw + ((size_t)b * L_ + lbase + lr) * C_;
#pragma unroll
    for (int kc = 0; kc < 4; ++kc)
      qfrag[kc] = combine_a(
          *reinterpret_cast<const h8*>(qrow + kc * 32 + 16 * half),
          *reinterpret_cast<const h8*>(qrow + kc * 32 + 16 * half + 8));
  }

  float mrow[8], lrow[8];
  v8f oacc[8];
#pragma unroll
  for (int i = 0; i < 8; ++i) { mrow[i] = NEG_INF; lrow[i] = 0.f; oacc[i] = v8f{}; }

  const float sl = SM_SCALE * LOG2E;  // work in exp2 domain

  for (int t = 0; t < L_ / 32; ++t) {
    const int kv0 = t * 32;
    const _Float16* krow0 = Kw + ((size_t)b * L_ + kv0 + lr) * C_;  // S cols 0-15
    const _Float16* krow1 = krow0 + 16 * C_;                        // S cols 16-31
    __builtin_prefetch(krow0 + 32 * C_, 0, 0);  // next kv tile hint

    // S = Q K^T : two 16x16 tiles, B-frag = 16 contiguous f16 per lane
    v8f s0 = {}, s1 = {};
#pragma unroll
    for (int kc = 0; kc < 4; ++kc) {
      const v16h bk0 =
          *reinterpret_cast<const v16h*>(krow0 + kc * 32 + 16 * half);
      const v16h bk1 =
          *reinterpret_cast<const v16h*>(krow1 + kc * 32 + 16 * half);
      s0 = wmma_f16(qfrag[kc], bk0, s0);
      s1 = wmma_f16(qfrag[kc], bk1, s1);
    }

    // Online softmax. Row i lives on VGPR i across the 16 lanes of a half.
    float corr[8];
#pragma unroll
    for (int i = 0; i < 8; ++i) {
      float a = s0[i] * sl, c = s1[i] * sl;
      float rm = fmaxf(a, c);
#pragma unroll
      for (int m = 1; m < 16; m <<= 1) rm = fmaxf(rm, __shfl_xor(rm, m, 16));
      const float mnew = fmaxf(mrow[i], rm);
      const float cr = exp2f(mrow[i] - mnew);
      const float p0 = exp2f(a - mnew), p1 = exp2f(c - mnew);
      float rs = p0 + p1;
#pragma unroll
      for (int m = 1; m < 16; m <<= 1) rs += __shfl_xor(rs, m, 16);
      lrow[i] = lrow[i] * cr + rs;
      mrow[i] = mnew;
      corr[i] = cr;
      s0[i] = p0; s1[i] = p1;
    }

    // Re-layout P (C/D layout -> A layout) via per-wave LDS tile.
#pragma unroll
    for (int i = 0; i < 8; ++i) {
      const int row = i + 8 * half;
      lds_p[wave][row][lr]      = (_Float16)s0[i];
      lds_p[wave][row][lr + 16] = (_Float16)s1[i];
    }
    // Intra-wave DS ops are in-order; pin scheduling + drain DScnt.
    asm volatile("s_wait_dscnt 0x0" ::: "memory");
    const _Float16* prow = &lds_p[wave][lr][0];
    const v16h pfrag = combine_a(
        *reinterpret_cast<const h8*>(prow + 16 * half),
        *reinterpret_cast<const h8*>(prow + 16 * half + 8));

    // O = diag(corr)*O + P V ; V B-frags are contiguous thanks to [B,C,L]
    const _Float16* vbase = Vw + (size_t)b * C_ * L_ + kv0 + 16 * half;
#pragma unroll
    for (int nt = 0; nt < 8; ++nt) {
      const v16h bv =
          *reinterpret_cast<const v16h*>(vbase + (size_t)(nt * 16 + lr) * L_);
      v8f oa = oacc[nt];
#pragma unroll
      for (int i = 0; i < 8; ++i) oa[i] *= corr[i];
      oacc[nt] = wmma_f16(pfrag, bv, oa);
    }
  }

  // Epilogue: normalize by row sums, ReLU, store f16 [B,L,C]
#pragma unroll
  for (int i = 0; i < 8; ++i) lrow[i] = 1.0f / lrow[i];
#pragma unroll
  for (int nt = 0; nt < 8; ++nt) {
#pragma unroll
    for (int i = 0; i < 8; ++i) {
      const float r = fmaxf(oacc[nt][i] * lrow[i], 0.0f);
      const int row = lbase + i + 8 * half;
      Rw[((size_t)b * L_ + row) * C_ + nt * 16 + lr] = (_Float16)r;
    }
  }
}

// ---------------------------------------------------------------------------
// Kernel 3: out = x + Wo * relu(r) + bo, out [B,C,L] f32 with residual.
// ---------------------------------------------------------------------------
__global__ __launch_bounds__(128) void out_proj_kernel(
    const _Float16* __restrict__ Rw, const _Float16* __restrict__ Woh,
    const float* __restrict__ bo, const float* __restrict__ x,
    float* __restrict__ out) {
  const int tid  = threadIdx.x;
  const int wave = tid >> 5, lane = tid & 31;
  const int half = lane >> 4, lr = lane & 15;
  const int blk  = blockIdx.x;
  const int b    = blk >> 5;
  const int lbase = (blk & 31) * 64 + wave * 16;

  v16h afrag[4];
  {
    const _Float16* rrow = Rw + ((size_t)b * L_ + lbase + lr) * C_;
#pragma unroll
    for (int kc = 0; kc < 4; ++kc)
      afrag[kc] = combine_a(
          *reinterpret_cast<const h8*>(rrow + kc * 32 + 16 * half),
          *reinterpret_cast<const h8*>(rrow + kc * 32 + 16 * half + 8));
  }

#pragma unroll 1
  for (int nt = 0; nt < 8; ++nt) {
    const int o = nt * 16 + lr;
    v8f acc = {};
#pragma unroll
    for (int kc = 0; kc < 4; ++kc) {
      const v16h bw = load_wfrag_h(Woh, o, kc * 32 + 16 * half);
      acc = wmma_f16(afrag[kc], bw, acc);
    }
    const float bias = bo[o];
    // Per lane: 8 consecutive l positions -> vectorized residual add + store
    const size_t base = ((size_t)b * C_ + o) * L_ + lbase + 8 * half;
    const float4 x0 = *reinterpret_cast<const float4*>(x + base);
    const float4 x1 = *reinterpret_cast<const float4*>(x + base + 4);
    float4 r0, r1;
    r0.x = acc[0] + bias + x0.x;  r0.y = acc[1] + bias + x0.y;
    r0.z = acc[2] + bias + x0.z;  r0.w = acc[3] + bias + x0.w;
    r1.x = acc[4] + bias + x1.x;  r1.y = acc[5] + bias + x1.y;
    r1.z = acc[6] + bias + x1.z;  r1.w = acc[7] + bias + x1.w;
    *reinterpret_cast<float4*>(out + base)     = r0;
    *reinterpret_cast<float4*>(out + base + 4) = r1;
  }
}

// ---------------------------------------------------------------------------
extern "C" void kernel_launch(void* const* d_in, const int* in_sizes, int n_in,
                              void* d_out, int out_size, void* d_ws,
                              size_t ws_size, hipStream_t stream) {
  const float* x  = (const float*)d_in[0];
  const float* Wq = (const float*)d_in[1];
  const float* bq = (const float*)d_in[2];
  const float* Wk = (const float*)d_in[3];
  const float* bk = (const float*)d_in[4];
  const float* Wv = (const float*)d_in[5];
  const float* bv = (const float*)d_in[6];
  const float* Wo = (const float*)d_in[7];
  const float* bo = (const float*)d_in[8];
  float* out = (float*)d_out;

  char* ws = (char*)d_ws;
  const size_t elems = (size_t)B_ * L_ * C_;        // 2Mi elements
  _Float16* Qw = (_Float16*)(ws);                   // 4 MiB each
  _Float16* Kw = (_Float16*)(ws + 2 * elems);
  _Float16* Vw = (_Float16*)(ws + 4 * elems);       // transposed [B,C,L]
  _Float16* Rw = (_Float16*)(ws + 6 * elems);
  _Float16* Wh = (_Float16*)(ws + 8 * elems);       // 4 x 32KiB f16 weights
  _Float16* Wqh = Wh + 0 * C_ * C_;
  _Float16* Wkh = Wh + 1 * C_ * C_;
  _Float16* Wvh = Wh + 2 * C_ * C_;
  _Float16* Woh = Wh + 3 * C_ * C_;

  const dim3 grid(B_ * 32);   // 32 l-tiles (64 rows) per batch
  const dim3 block(128);      // 4 waves, 16 rows each

  wcvt_kernel<<<dim3(16, 4), 256, 0, stream>>>(Wq, Wk, Wv, Wo, Wh);
  qkv_proj_kernel<<<grid, block, 0, stream>>>(x, Wqh, bq, Wkh, bk, Wvh, bv,
                                              Qw, Kw, Vw);
  attn_kernel<<<grid, block, 0, stream>>>(Qw, Kw, Vw, Rw);
  out_proj_kernel<<<grid, block, 0, stream>>>(Rw, Woh, bo, x, out);
}
